// QuantLinear8bit_73718818668911
// MI455X (gfx1250) — matrix operand order repaired and checked by reading, not compiled, not run on hardware
//
#include <hip/hip_runtime.h>

// ---- CDNA5 WMMA vector types ----
typedef __attribute__((ext_vector_type(16))) _Float16 v16h;
typedef __attribute__((ext_vector_type(8)))  _Float16 v8h;
typedef __attribute__((ext_vector_type(4)))  _Float16 v4h;
typedef __attribute__((ext_vector_type(8)))  float    v8f;

// Problem shape (fixed by the reference)
constexpr int M_TOT = 4 * 2048;   // B*S = 8192
constexpr int K_TOT = 4096;       // IN
constexpr int N_TOT = 11008;      // OUT

// ---- fast-path tiling (async kernel): 128x256 block, 64x64 per wave ----
constexpr int ABM = 128;
constexpr int ABN = 256;          // 11008 = 43 * 256
constexpr int ABK = 32;
constexpr int ALD = ABK + 8;      // LDS row stride (halves): 80 B
constexpr int AKT = K_TOT / ABK;  // 128 k-tiles

// ---- fallback tiling: 128x128 block, 64x32 per wave ----
constexpr int BM  = 128;
constexpr int BN  = 128;
constexpr int BK  = 32;
constexpr int LDT = BK + 8;
constexpr int KT  = K_TOT / BK;

// ---------------------------------------------------------------------------
// CDNA5 async global->LDS copy (ASYNCcnt path), 16 bytes per lane.
// VDST = 32-bit LDS byte address VGPR, VADDR = 64-bit global address.
// ---------------------------------------------------------------------------
__device__ __forceinline__ unsigned lds_off(const void* p) {
    // Generic LDS addresses are {SHARED_BASE, offset}: low 32 bits are the
    // LDS byte address the DS/async hardware consumes (ISA 10.2 aperture map).
    return (unsigned)(uintptr_t)p;
}

__device__ __forceinline__ void async_cp16(unsigned lds, const _Float16* g) {
    asm volatile("global_load_async_to_lds_b128 %0, %1, off"
                 :: "v"(lds), "v"(g) : "memory");
}

__device__ __forceinline__ void wait_async0() {
    asm volatile("s_wait_asynccnt 0" ::: "memory");
}

// ---------------------------------------------------------------------------
// Pre-conversion kernels (run once per call; output lives in d_ws)
// ---------------------------------------------------------------------------
__global__ __launch_bounds__(256)
void convert_w_kernel(const int* __restrict__ q, _Float16* __restrict__ w) {
    const size_t i = ((size_t)blockIdx.x * 256 + threadIdx.x) * 8;
    int4 a = *reinterpret_cast<const int4*>(q + i);
    int4 b = *reinterpret_cast<const int4*>(q + i + 4);
    v4h h0 = { (_Float16)a.x, (_Float16)a.y, (_Float16)a.z, (_Float16)a.w };
    v4h h1 = { (_Float16)b.x, (_Float16)b.y, (_Float16)b.z, (_Float16)b.w };
    *reinterpret_cast<v4h*>(w + i)     = h0;   // int8 values are exact in f16
    *reinterpret_cast<v4h*>(w + i + 4) = h1;
}

__global__ __launch_bounds__(256)
void convert_x_kernel(const float* __restrict__ x, _Float16* __restrict__ xh) {
    const size_t i = ((size_t)blockIdx.x * 256 + threadIdx.x) * 8;
    float4 a = *reinterpret_cast<const float4*>(x + i);
    float4 b = *reinterpret_cast<const float4*>(x + i + 4);
    v4h h0 = { (_Float16)a.x, (_Float16)a.y, (_Float16)a.z, (_Float16)a.w };
    v4h h1 = { (_Float16)b.x, (_Float16)b.y, (_Float16)b.z, (_Float16)b.w };
    *reinterpret_cast<v4h*>(xh + i)     = h0;
    *reinterpret_cast<v4h*>(xh + i + 4) = h1;
}

// ---------------------------------------------------------------------------
// Fast path: f16 WMMA GEMM, 128x256 block, async global->LDS double buffering
// ---------------------------------------------------------------------------
__global__ __launch_bounds__(256)
void qlinear_wmma_async(const _Float16* __restrict__ Xh,
                        const _Float16* __restrict__ Wh,
                        const float* __restrict__ Scale,
                        const float* __restrict__ Bias,
                        float* __restrict__ Out)
{
    __shared__ __attribute__((aligned(16))) _Float16 As[2][ABM * ALD]; // 2*10 KB
    __shared__ __attribute__((aligned(16))) _Float16 Bs[2][ABN * ALD]; // 2*20 KB

    const int tid   = threadIdx.x;
    const int lane  = tid & 31;
    const int wave  = tid >> 5;
    const int waveM = wave & 1;       // 2 waves along M -> 64 rows each
    const int waveN = wave >> 1;      // 4 waves along N -> 64 cols each

    const int m0 = blockIdx.y * ABM;
    const int n0 = blockIdx.x * ABN;

    // ---- async staging: 16 B chunks; A tile = 512 chunks (2/thread),
    //                                  B tile = 1024 chunks (4/thread).
    const _Float16* gA[2];
    const _Float16* gB[4];
    unsigned lA[2][2], lB[4][2];

    #pragma unroll
    for (int j = 0; j < 2; ++j) {
        const int c   = tid + 256 * j;
        const int row = c >> 2, col = (c & 3) * 8;
        gA[j]    = Xh + (size_t)(m0 + row) * K_TOT + col;
        lA[j][0] = lds_off(&As[0][row * ALD + col]);
        lA[j][1] = lds_off(&As[1][row * ALD + col]);
    }
    #pragma unroll
    for (int j = 0; j < 4; ++j) {
        const int c   = tid + 256 * j;
        const int row = c >> 2, col = (c & 3) * 8;
        gB[j]    = Wh + (size_t)(n0 + row) * K_TOT + col;
        lB[j][0] = lds_off(&Bs[0][row * ALD + col]);
        lB[j][1] = lds_off(&Bs[1][row * ALD + col]);
    }

    // ---- WMMA fragment lane addressing (ISA 16-bit A 16x32 / B 32x16 layouts)
    const int aRow = lane & 15;
    const int aK   = (lane >> 4) * 8;
    const int bCol = lane & 15;
    const int bK   = (lane >> 4) * 8;

    v8f acc[4][4] = {};   // 4 M-tiles x 4 N-tiles of 16x16 f32

    auto issue = [&](int b) {
        #pragma unroll
        for (int j = 0; j < 2; ++j) { async_cp16(lA[j][b], gA[j]); gA[j] += ABK; }
        #pragma unroll
        for (int j = 0; j < 4; ++j) { async_cp16(lB[j][b], gB[j]); gB[j] += ABK; }
    };

    auto compute = [&](const _Float16* __restrict__ as,
                       const _Float16* __restrict__ bs) {
        v16h a[4];
        #pragma unroll
        for (int mt = 0; mt < 4; ++mt) {
            const int base = (waveM * 64 + mt * 16 + aRow) * ALD + aK;
            v8h lo = *reinterpret_cast<const v8h*>(as + base);       // K = aK..aK+7
            v8h hi = *reinterpret_cast<const v8h*>(as + base + 16);  // K = aK+16..aK+23
            a[mt] = __builtin_shufflevector(lo, hi, 0,1,2,3,4,5,6,7,8,9,10,11,12,13,14,15);
        }
        #pragma unroll
        for (int nt = 0; nt < 4; ++nt) {
            const int base = (waveN * 64 + nt * 16 + bCol) * ALD + bK;
            v8h lo = *reinterpret_cast<const v8h*>(bs + base);       // K = bK..bK+7
            v8h hi = *reinterpret_cast<const v8h*>(bs + base + 16);  // K = bK+16..bK+23
            v16h b = __builtin_shufflevector(lo, hi, 0,1,2,3,4,5,6,7,8,9,10,11,12,13,14,15);
            #pragma unroll
            for (int mt = 0; mt < 4; ++mt)
                acc[mt][nt] = __builtin_amdgcn_wmma_f32_16x16x32_f16(
                    false, a[mt], false, b, (short)0, acc[mt][nt], false, false);
        }
    };

    // ---- pipeline: tile0 -> buf0
    issue(0);
    wait_async0();
    __syncthreads();

    // Unrolled by 2 so buffer parity (and LDS addresses) are loop-invariant.
    for (int kt = 0; kt < AKT; kt += 2) {
        issue(1);                       // prefetch tile kt+1 (kt+1 <= AKT-1 always)
        compute(As[0], Bs[0]);          // consume tile kt
        wait_async0();
        __syncthreads();

        if (kt + 2 < AKT) issue(0);     // prefetch tile kt+2
        compute(As[1], Bs[1]);          // consume tile kt+1
        wait_async0();
        __syncthreads();
    }

    // ---- epilogue: per-output-channel dequant scale + bias, fp32 store
    #pragma unroll
    for (int nt = 0; nt < 4; ++nt) {
        const int   col = n0 + waveN * 64 + nt * 16 + bCol;
        const float s   = Scale[col];
        const float bz  = Bias[col];
        #pragma unroll
        for (int mt = 0; mt < 4; ++mt) {
            const int rowb = m0 + waveM * 64 + mt * 16 + (lane >> 4) * 8;
            #pragma unroll
            for (int i = 0; i < 8; ++i)
                Out[(size_t)(rowb + i) * N_TOT + col] = acc[mt][nt][i] * s + bz;
        }
    }
}

// ---------------------------------------------------------------------------
// Fallback (round-1 kernel): fused conversion, no workspace needed
// ---------------------------------------------------------------------------
__global__ __launch_bounds__(256)
void qlinear_wmma_fused(const float* __restrict__ X,
                        const int*   __restrict__ Wq,
                        const float* __restrict__ Scale,
                        const float* __restrict__ Bias,
                        float* __restrict__ Out)
{
    __shared__ __attribute__((aligned(16))) _Float16 As[2][BM * LDT];
    __shared__ __attribute__((aligned(16))) _Float16 Bs[2][BN * LDT];

    const int tid   = threadIdx.x;
    const int lane  = tid & 31;
    const int wave  = tid >> 5;
    const int waveM = wave & 1;
    const int waveN = wave >> 1;

    const int m0 = blockIdx.y * BM;
    const int n0 = blockIdx.x * BN;

    const int sRow = tid >> 3;
    const int sCol = (tid & 7) * 4;

    const int aRow = lane & 15;
    const int aK   = (lane >> 4) * 8;
    const int bCol = lane & 15;
    const int bK   = (lane >> 4) * 8;

    float4 xr[4];
    int4   wr[4];
    v8f acc[4][2] = {};

    {
        #pragma unroll
        for (int p = 0; p < 4; ++p) {
            const int r = p * 32 + sRow;
            xr[p] = *reinterpret_cast<const float4*>(X  + (size_t)(m0 + r) * K_TOT + sCol);
            wr[p] = *reinterpret_cast<const int4*>  (Wq + (size_t)(n0 + r) * K_TOT + sCol);
        }
        #pragma unroll
        for (int p = 0; p < 4; ++p) {
            const int r = p * 32 + sRow;
            v4h ah = { (_Float16)xr[p].x, (_Float16)xr[p].y,
                       (_Float16)xr[p].z, (_Float16)xr[p].w };
            *reinterpret_cast<v4h*>(&As[0][r * LDT + sCol]) = ah;
            v4h bh = { (_Float16)wr[p].x, (_Float16)wr[p].y,
                       (_Float16)wr[p].z, (_Float16)wr[p].w };
            *reinterpret_cast<v4h*>(&Bs[0][r * LDT + sCol]) = bh;
        }
    }
    __syncthreads();

    for (int kt = 0; kt < KT; ++kt) {
        const int  cur  = kt & 1;
        const bool more = (kt + 1) < KT;

        if (more) {
            const int k0 = (kt + 1) * BK;
            #pragma unroll
            for (int p = 0; p < 4; ++p) {
                const int r = p * 32 + sRow;
                xr[p] = *reinterpret_cast<const float4*>(X  + (size_t)(m0 + r) * K_TOT + k0 + sCol);
                wr[p] = *reinterpret_cast<const int4*>  (Wq + (size_t)(n0 + r) * K_TOT + k0 + sCol);
            }
        }

        const _Float16* __restrict__ as = As[cur];
        const _Float16* __restrict__ bs = Bs[cur];

        v16h a[4], b[2];
        #pragma unroll
        for (int mt = 0; mt < 4; ++mt) {
            const int base = (waveM * 64 + mt * 16 + aRow) * LDT + aK;
            v8h lo = *reinterpret_cast<const v8h*>(as + base);
            v8h hi = *reinterpret_cast<const v8h*>(as + base + 16);
            a[mt] = __builtin_shufflevector(lo, hi, 0,1,2,3,4,5,6,7,8,9,10,11,12,13,14,15);
        }
        #pragma unroll
        for (int nt = 0; nt < 2; ++nt) {
            const int base = (waveN * 32 + nt * 16 + bCol) * LDT + bK;
            v8h lo = *reinterpret_cast<const v8h*>(bs + base);
            v8h hi = *reinterpret_cast<const v8h*>(bs + base + 16);
            b[nt] = __builtin_shufflevector(lo, hi, 0,1,2,3,4,5,6,7,8,9,10,11,12,13,14,15);
        }

        #pragma unroll
        for (int mt = 0; mt < 4; ++mt)
            #pragma unroll
            for (int nt = 0; nt < 2; ++nt)
                acc[mt][nt] = __builtin_amdgcn_wmma_f32_16x16x32_f16(
                    false, a[mt], false, b[nt], (short)0, acc[mt][nt], false, false);

        if (more) {
            const int nxt = (kt + 1) & 1;
            #pragma unroll
            for (int p = 0; p < 4; ++p) {
                const int r = p * 32 + sRow;
                v4h ah = { (_Float16)xr[p].x, (_Float16)xr[p].y,
                           (_Float16)xr[p].z, (_Float16)xr[p].w };
                *reinterpret_cast<v4h*>(&As[nxt][r * LDT + sCol]) = ah;
                v4h bh = { (_Float16)wr[p].x, (_Float16)wr[p].y,
                           (_Float16)wr[p].z, (_Float16)wr[p].w };
                *reinterpret_cast<v4h*>(&Bs[nxt][r * LDT + sCol]) = bh;
            }
        }
        __syncthreads();
    }

    #pragma unroll
    for (int nt = 0; nt < 2; ++nt) {
        const int   col = n0 + waveN * 32 + nt * 16 + bCol;
        const float s   = Scale[col];
        const float bz  = Bias[col];
        #pragma unroll
        for (int mt = 0; mt < 4; ++mt) {
            const int rowb = m0 + waveM * 64 + mt * 16 + (lane >> 4) * 8;
            #pragma unroll
            for (int i = 0; i < 8; ++i)
                Out[(size_t)(rowb + i) * N_TOT + col] = acc[mt][nt][i] * s + bz;
        }
    }
}

// ---------------------------------------------------------------------------
extern "C" void kernel_launch(void* const* d_in, const int* in_sizes, int n_in,
                              void* d_out, int out_size, void* d_ws, size_t ws_size,
                              hipStream_t stream) {
    (void)in_sizes; (void)n_in; (void)out_size;
    const float* x     = (const float*)d_in[0];
    const int*   qw    = (const int*)  d_in[1];
    const float* scale = (const float*)d_in[2];
    const float* bias  = (const float*)d_in[3];
    float*       out   = (float*)d_out;

    const size_t wBytes = (size_t)N_TOT * K_TOT * sizeof(_Float16); // 90.2 MB
    const size_t xBytes = (size_t)M_TOT * K_TOT * sizeof(_Float16); // 67.1 MB

    dim3 block(256);

    if (d_ws != nullptr && ws_size >= wBytes + xBytes) {
        _Float16* Wh = (_Float16*)d_ws;
        _Float16* Xh = (_Float16*)((char*)d_ws + wBytes);
        const int wBlocks = (int)(((size_t)N_TOT * K_TOT) / (256 * 8)); // 22016
        const int xBlocks = (int)(((size_t)M_TOT * K_TOT) / (256 * 8)); // 16384
        convert_w_kernel<<<wBlocks, block, 0, stream>>>(qw, Wh);
        convert_x_kernel<<<xBlocks, block, 0, stream>>>(x, Xh);
        dim3 grid(N_TOT / ABN, M_TOT / ABM);  // (43, 64)
        qlinear_wmma_async<<<grid, block, 0, stream>>>(Xh, Wh, scale, bias, out);
    } else {
        dim3 grid(N_TOT / BN, M_TOT / BM);    // (86, 64)
        qlinear_wmma_fused<<<grid, block, 0, stream>>>(x, qw, scale, bias, out);
    }
}
